// Rel_Embedding_17205638988000
// MI455X (gfx1250) — compile-verified, matrix operand order checked by reading, not run on hardware
//
#include <hip/hip_runtime.h>
#include <hip/hip_bf16.h>

// ---------------------------------------------------------------------------
// CDNA5 (gfx1250) fused Rel_Embedding forward.
//   g, sel, sel_logits = ref(x, rel_x, W1..b5)
// bf16 WMMA (v_wmma_f32_16x16x32_bf16), wave32, 8 waves / block.
// ---------------------------------------------------------------------------

typedef __attribute__((ext_vector_type(16))) __bf16        v16bf;
typedef __attribute__((ext_vector_type(8)))  float         v8f;
typedef __attribute__((ext_vector_type(4)))  unsigned int  v4u;

#define DINLINE static __device__ __forceinline__

// --- workspace byte layout (all 16B aligned) -------------------------------
static constexpr size_t REL_AGG_B = 0;                         // 6144*64 f32
static constexpr size_t W1T_B = 6144ull * 64 * 4;              // 1,572,864
static constexpr size_t W2T_B = W1T_B + 128 * 96 * 2;          // +24,576
static constexpr size_t W3T_B = W2T_B + 80 * 128 * 2;          // +20,480
static constexpr size_t W4T_B = W3T_B + 256 * 96 * 2;          // +49,152
static constexpr size_t W5T_B = W4T_B + 256 * 256 * 2;         // +131,072
static constexpr size_t WS_NEEDED = W5T_B + 64 * 256 * 2;      // ~1.83 MB

// --- helpers ---------------------------------------------------------------
DINLINE unsigned short f32_bf16(float f) {                     // RNE
  unsigned int u = __float_as_uint(f);
  u += 0x7FFFu + ((u >> 16) & 1u);
  return (unsigned short)(u >> 16);
}
DINLINE unsigned int pack_bf16x2(float a, float b) {
  return (unsigned int)f32_bf16(a) | ((unsigned int)f32_bf16(b) << 16);
}
DINLINE v8f zero8() { v8f z = {0.f,0.f,0.f,0.f,0.f,0.f,0.f,0.f}; return z; }

// 16x32 bf16 fragment load (A: rc=row base / B from NxK-major: rc=col base).
// ISA layout: lanes 0-15 hold K{k0..k0+7 , k0+16..k0+23}, lanes 16-31 shift +8.
DINLINE v16bf load_frag(const unsigned short* base, int ldk, int rc, int kt,
                        int lane) {
  const int r  = rc + (lane & 15);
  const int k0 = kt + ((lane >> 4) << 3);
  union { v4u q[2]; v16bf v; } f;
  f.q[0] = *(const v4u*)(base + r * ldk + k0);
  f.q[1] = *(const v4u*)(base + r * ldk + k0 + 16);
  return f.v;
}

DINLINE v8f wmma_bf16(v16bf a, v16bf b, v8f c) {
  return __builtin_amdgcn_wmma_f32_16x16x32_bf16(
      /*neg_a=*/false, a, /*neg_b=*/false, b,
      /*c_mod=*/(short)0, c, /*reuse_a=*/false, /*reuse_b=*/false);
}

// ---------------------------------------------------------------------------
// Kernel 0: transpose+convert weights to bf16 N-major in ws.
// ---------------------------------------------------------------------------
__global__ void prep_weights(const float* __restrict__ W1,
                             const float* __restrict__ W2,
                             const float* __restrict__ W3,
                             const float* __restrict__ W4,
                             const float* __restrict__ W5,
                             unsigned short* __restrict__ w1t,
                             unsigned short* __restrict__ w2t,
                             unsigned short* __restrict__ w3t,
                             unsigned short* __restrict__ w4t,
                             unsigned short* __restrict__ w5t) {
  int idx = blockIdx.x * 256 + threadIdx.x;
  if (idx < 128 * 96) {                       // W1T [128][96]
    int n = idx / 96, k = idx % 96;
    w1t[idx] = f32_bf16(W1[k * 128 + n]);
    return;
  }
  idx -= 128 * 96;
  if (idx < 80 * 128) {                       // W2T [80][128], cols 65..79 = 0
    int n = idx / 128, k = idx % 128;
    w2t[idx] = (n < 65) ? f32_bf16(W2[k * 65 + n]) : (unsigned short)0;
    return;
  }
  idx -= 80 * 128;
  if (idx < 256 * 96) {                       // W3T [256][96]
    int n = idx / 96, k = idx % 96;
    w3t[idx] = f32_bf16(W3[k * 256 + n]);
    return;
  }
  idx -= 256 * 96;
  if (idx < 256 * 256) {                      // W4T [256][256]
    int n = idx / 256, k = idx % 256;
    w4t[idx] = f32_bf16(W4[k * 256 + n]);
    return;
  }
  idx -= 256 * 256;
  if (idx < 64 * 256) {                       // W5T [64][256]
    int n = idx / 256, k = idx % 256;
    w5t[idx] = f32_bf16(W5[k * 64 + n]);
  }
}

// ---------------------------------------------------------------------------
// Kernel 1: per-(b,i) fused relation MLP + epilogue.  6144 blocks x 256 thr.
//   A = rel_x rows [pi*96, pi*96+96)  (96x96)
//   h = relu(A @ W1 + b1)             (96x128)  -> LDS bf16
//   o = h @ W2 + b2                   (96x80)   -> LDS f32 (col64 = logit)
//   sel/logits out; rel_agg[pi][c] = sum_j relu(o[j][c])*sel[j]
// ---------------------------------------------------------------------------
__global__ __launch_bounds__(256)
void rel_mlp_kernel(const float* __restrict__ rel_x,
                    const float* __restrict__ b1,
                    const float* __restrict__ b2,
                    const unsigned short* __restrict__ w1t,
                    const unsigned short* __restrict__ w2t,
                    float* __restrict__ out_sel,
                    float* __restrict__ out_logits,
                    float* __restrict__ rel_agg) {
  constexpr int LDA = 104;   // bf16 stride for A tile   (96 rows x 96 K)
  constexpr int LDH = 136;   // bf16 stride for h        (96 rows x 128 K)
  constexpr int LDO = 88;    // f32 stride for o         (96 rows x 80 cols)
  __shared__ __align__(16) float smemf[96 * LDO];   // 33,792 B, phase-reused
  __shared__ float selbuf[96];

  unsigned short* Abuf = (unsigned short*)smemf;
  unsigned short* Hbuf = (unsigned short*)smemf;
  float*          Obuf = smemf;

  const int tid    = threadIdx.x;
  const int lane   = tid & 31;
  const int wv     = tid >> 5;
  const int pi     = blockIdx.x;        // = b*96 + i
  const int i_in_n = pi % 96;

  // --- stage A tile: 96x96 f32 -> bf16 LDS (rows never cross float4) ------
  const float* Ag = rel_x + (size_t)pi * (96 * 96);
  for (int c = tid; c < (96 * 96) / 4; c += 256) {
    const int p = c * 4;
    const int row = p / 96, col = p % 96;
    const float4 f = *(const float4*)(Ag + p);
    *(unsigned int*)(Abuf + row * LDA + col)     = pack_bf16x2(f.x, f.y);
    *(unsigned int*)(Abuf + row * LDA + col + 2) = pack_bf16x2(f.z, f.w);
  }
  __syncthreads();

  // --- GEMM1: wave wv owns h cols [16wv, 16wv+16) --------------------------
  v8f acc1[6];
  for (int m = 0; m < 6; ++m) acc1[m] = zero8();
  {
    const int ncol = wv << 4;
    for (int kk = 0; kk < 3; ++kk) {
      const v16bf bfrag = load_frag(w1t, 96, ncol, kk * 32, lane);
      for (int m = 0; m < 6; ++m) {
        const v16bf afrag = load_frag(Abuf, LDA, m * 16, kk * 32, lane);
        acc1[m] = wmma_bf16(afrag, bfrag, acc1[m]);
      }
    }
  }
  __syncthreads();                       // A reads done; reuse buffer for h

  // --- h = relu(acc1 + b1) -> LDS bf16 ------------------------------------
  {
    const int col   = (wv << 4) + (lane & 15);
    const float bia = b1[col];
    const int rbase = (lane >> 4) << 3;
    for (int m = 0; m < 6; ++m)
      for (int r = 0; r < 8; ++r)
        Hbuf[(m * 16 + rbase + r) * LDH + col] =
            f32_bf16(fmaxf(acc1[m][r] + bia, 0.f));
  }
  __syncthreads();

  // --- GEMM2: waves 0..4 own o cols [16wv, 16wv+16) ------------------------
  v8f acc2[6];
  for (int m = 0; m < 6; ++m) acc2[m] = zero8();
  if (wv < 5) {
    const int ncol = wv << 4;
    for (int kk = 0; kk < 4; ++kk) {
      const v16bf bfrag = load_frag(w2t, 128, ncol, kk * 32, lane);
      for (int m = 0; m < 6; ++m) {
        const v16bf afrag = load_frag(Hbuf, LDH, m * 16, kk * 32, lane);
        acc2[m] = wmma_bf16(afrag, bfrag, acc2[m]);
      }
    }
  }
  __syncthreads();                       // h reads done; reuse buffer for o

  // --- o = acc2 + b2 -> LDS f32 --------------------------------------------
  if (wv < 5) {
    const int col   = (wv << 4) + (lane & 15);
    const float bia = (col < 65) ? b2[col] : 0.f;
    const int rbase = (lane >> 4) << 3;
    for (int m = 0; m < 6; ++m)
      for (int r = 0; r < 8; ++r)
        Obuf[(m * 16 + rbase + r) * LDO + col] = acc2[m][r] + bia;
  }
  __syncthreads();

  // --- epilogue A: sigmoid / round-to-even / diag mask ---------------------
  if (tid < 96) {
    const int j = tid;
    const float logit = Obuf[j * LDO + 64];
    const float sig   = 1.f / (1.f + __expf(-logit));
    float sel         = rintf(sig);
    if (j == i_in_n) sel = 0.f;          // mask_I = 1 - eye(N)
    const size_t gr = (size_t)pi * 96 + j;
    out_logits[gr] = logit;              // pre-mask logits
    out_sel[gr]    = sel;                // post-mask sel
    selbuf[j]      = sel;
  }
  __syncthreads();

  // --- epilogue B: rel_agg[pi][c] = sum_j relu(o[j][c]) * sel[j] -----------
  if (tid < 64) {
    const int c = tid;
    float acc = 0.f;
    for (int j = 0; j < 96; ++j)
      acc += fmaxf(Obuf[j * LDO + c], 0.f) * selbuf[j];
    rel_agg[(size_t)pi * 64 + c] = acc;
  }
}

// ---------------------------------------------------------------------------
// Kernel 2: head MLP over 6144 rows.  64 blocks x 256 thr, 96 rows each.
//   a0 = concat(x, rel_agg)  (96x96)
//   h2 = relu(a0@W3+b3); h3 = relu(h2@W4+b4); g = h3@W5+b5
// ---------------------------------------------------------------------------
__global__ __launch_bounds__(256)
void head_mlp_kernel(const float* __restrict__ x,
                     const float* __restrict__ rel_agg,
                     const float* __restrict__ b3,
                     const float* __restrict__ b4,
                     const float* __restrict__ b5,
                     const unsigned short* __restrict__ w3t,
                     const unsigned short* __restrict__ w4t,
                     const unsigned short* __restrict__ w5t,
                     float* __restrict__ g) {
  constexpr int LDA = 104;   // a0 stride (bf16), 96x96
  constexpr int LDH = 264;   // h stride (bf16), 96x256
  __shared__ __align__(16) unsigned short smem[96 * LDH];   // 50,688 B
  unsigned short* Abuf = smem;
  unsigned short* Hbuf = smem;

  const int tid = threadIdx.x, lane = tid & 31, wv = tid >> 5;
  const int rb = blockIdx.x * 96;

  // --- stage a0 = [x | rel_agg] as bf16 ------------------------------------
  for (int c = tid; c < 96 * 32; c += 256) {
    const int row = c >> 5, col = c & 31;
    Abuf[row * LDA + col] = f32_bf16(x[(size_t)(rb + row) * 32 + col]);
  }
  for (int c = tid; c < 96 * 64; c += 256) {
    const int row = c / 64, col = c % 64;
    Abuf[row * LDA + 32 + col] =
        f32_bf16(rel_agg[(size_t)(rb + row) * 64 + col]);
  }
  __syncthreads();

  // --- GEMM3: wave wv owns cols [32wv, 32wv+32) (2 n-tiles), K=96 ----------
  v8f acc[2][6];
  for (int n = 0; n < 2; ++n)
    for (int m = 0; m < 6; ++m) acc[n][m] = zero8();
  for (int kk = 0; kk < 3; ++kk) {
    const v16bf bf0 = load_frag(w3t, 96, (wv << 5),      kk * 32, lane);
    const v16bf bf1 = load_frag(w3t, 96, (wv << 5) + 16, kk * 32, lane);
    for (int m = 0; m < 6; ++m) {
      const v16bf a = load_frag(Abuf, LDA, m * 16, kk * 32, lane);
      acc[0][m] = wmma_bf16(a, bf0, acc[0][m]);
      acc[1][m] = wmma_bf16(a, bf1, acc[1][m]);
    }
  }
  __syncthreads();
  {                                       // h2 = relu(acc + b3) -> LDS bf16
    const int rbase = (lane >> 4) << 3;
    for (int n = 0; n < 2; ++n) {
      const int col   = (wv << 5) + (n << 4) + (lane & 15);
      const float bia = b3[col];
      for (int m = 0; m < 6; ++m)
        for (int r = 0; r < 8; ++r)
          Hbuf[(m * 16 + rbase + r) * LDH + col] =
              f32_bf16(fmaxf(acc[n][m][r] + bia, 0.f));
    }
  }
  __syncthreads();

  // --- GEMM4: same tiling, K=256 -------------------------------------------
  for (int n = 0; n < 2; ++n)
    for (int m = 0; m < 6; ++m) acc[n][m] = zero8();
  for (int kk = 0; kk < 8; ++kk) {
    const v16bf bf0 = load_frag(w4t, 256, (wv << 5),      kk * 32, lane);
    const v16bf bf1 = load_frag(w4t, 256, (wv << 5) + 16, kk * 32, lane);
    for (int m = 0; m < 6; ++m) {
      const v16bf a = load_frag(Hbuf, LDH, m * 16, kk * 32, lane);
      acc[0][m] = wmma_bf16(a, bf0, acc[0][m]);
      acc[1][m] = wmma_bf16(a, bf1, acc[1][m]);
    }
  }
  __syncthreads();
  {                                       // h3 = relu(acc + b4) overwrites h2
    const int rbase = (lane >> 4) << 3;
    for (int n = 0; n < 2; ++n) {
      const int col   = (wv << 5) + (n << 4) + (lane & 15);
      const float bia = b4[col];
      for (int m = 0; m < 6; ++m)
        for (int r = 0; r < 8; ++r)
          Hbuf[(m * 16 + rbase + r) * LDH + col] =
              f32_bf16(fmaxf(acc[n][m][r] + bia, 0.f));
    }
  }
  __syncthreads();

  // --- GEMM5: waves 0..3 own g cols [16wv, 16wv+16), K=256 -----------------
  if (wv < 4) {
    v8f acc5[6];
    for (int m = 0; m < 6; ++m) acc5[m] = zero8();
    for (int kk = 0; kk < 8; ++kk) {
      const v16bf bf = load_frag(w5t, 256, (wv << 4), kk * 32, lane);
      for (int m = 0; m < 6; ++m) {
        const v16bf a = load_frag(Hbuf, LDH, m * 16, kk * 32, lane);
        acc5[m] = wmma_bf16(a, bf, acc5[m]);
      }
    }
    const int col   = (wv << 4) + (lane & 15);
    const float bia = b5[col];
    const int rbase = (lane >> 4) << 3;
    for (int m = 0; m < 6; ++m)
      for (int r = 0; r < 8; ++r) {
        const int row = rb + m * 16 + rbase + r;
        g[(size_t)row * 64 + col] = acc5[m][r] + bia;
      }
  }
}

// ---------------------------------------------------------------------------
extern "C" void kernel_launch(void* const* d_in, const int* in_sizes, int n_in,
                              void* d_out, int out_size, void* d_ws,
                              size_t ws_size, hipStream_t stream) {
  (void)in_sizes; (void)n_in; (void)out_size;
  if (ws_size < WS_NEEDED) return;

  const float* x     = (const float*)d_in[0];
  const float* rel_x = (const float*)d_in[1];
  const float* W1 = (const float*)d_in[2];
  const float* b1 = (const float*)d_in[3];
  const float* W2 = (const float*)d_in[4];
  const float* b2 = (const float*)d_in[5];
  const float* W3 = (const float*)d_in[6];
  const float* b3 = (const float*)d_in[7];
  const float* W4 = (const float*)d_in[8];
  const float* b4 = (const float*)d_in[9];
  const float* W5 = (const float*)d_in[10];
  const float* b5 = (const float*)d_in[11];

  unsigned char* ws = (unsigned char*)d_ws;
  float*          rel_agg = (float*)(ws + REL_AGG_B);
  unsigned short* w1t = (unsigned short*)(ws + W1T_B);
  unsigned short* w2t = (unsigned short*)(ws + W2T_B);
  unsigned short* w3t = (unsigned short*)(ws + W3T_B);
  unsigned short* w4t = (unsigned short*)(ws + W4T_B);
  unsigned short* w5t = (unsigned short*)(ws + W5T_B);

  float* out_g      = (float*)d_out;                 // 64*96*64
  float* out_sel    = out_g + 64 * 96 * 64;          // 64*9216
  float* out_logits = out_sel + 64 * 9216;           // 64*9216

  prep_weights<<<504, 256, 0, stream>>>(W1, W2, W3, W4, W5,
                                        w1t, w2t, w3t, w4t, w5t);
  rel_mlp_kernel<<<6144, 256, 0, stream>>>(rel_x, b1, b2, w1t, w2t,
                                           out_sel, out_logits, rel_agg);
  head_mlp_kernel<<<64, 256, 0, stream>>>(x, rel_agg, b3, b4, b5,
                                          w3t, w4t, w5t, out_g);
}
